// MultiHeadSelfAttention_39779987096303
// MI455X (gfx1250) — matrix operand order repaired and checked by reading, not compiled, not run on hardware
//
#include <hip/hip_runtime.h>
#include <hip/hip_bf16.h>
#include <stdint.h>

// ---------------------------------------------------------------------------
// Multi-head causal self-attention for MI455X (gfx1250, wave32, WMMA bf16).
//   B=4, T=2048, C_IN=1024, H=16, Dh=64, D=1024
// GEMM-shaped work -> v_wmma_f32_16x16x32_bf16.
// Tile staging -> global_load_async_to_lds_b128 (ASYNCcnt) where no transpose
// is needed; global_prefetch_b8 for the next K/V tiles in the flash loop.
// ---------------------------------------------------------------------------

typedef __bf16 bf16;
typedef __attribute__((ext_vector_type(16))) __bf16 v16bf;
typedef __attribute__((ext_vector_type(8)))  float    v8f;

typedef int v4i __attribute__((vector_size(16)));
typedef __attribute__((address_space(1))) v4i gv4i;   // global int4
typedef __attribute__((address_space(3))) v4i lv4i;   // LDS int4

#define WMMA_BF16(a, b, c) \
  __builtin_amdgcn_wmma_f32_16x16x32_bf16(false, (a), false, (b), (short)0, (c), false, false)

// Async 16B global->LDS copy (per-lane addresses), tracked by ASYNCcnt.
#define ASYNC_LDS_B128(gp, lp) \
  __builtin_amdgcn_global_load_async_to_lds_b128((gv4i*)(gp), (lv4i*)(lp), 0, 0)

__device__ __forceinline__ bf16 f32_to_bf16(float f) {
  union { float f; uint32_t u; } x; x.f = f;
  uint32_t r = x.u + 0x7FFFu + ((x.u >> 16) & 1u);   // round-to-nearest-even
  union { uint16_t u; bf16 b; } y; y.u = (uint16_t)(r >> 16);
  return y.b;
}

union FragU { v16bf v; uint4 q[2]; };

// A-fragment (16x32, row per lane): element e holds K = ((e&8)?e+8:e) + 8*half.
// -> two contiguous 8-elem (16B) runs at +8*half and +8*half+16.
__device__ __forceinline__ v16bf frag_a_load(const bf16* p, int half) {
  FragU f;
  const bf16* b = p + half * 8;
  f.q[0] = *(const uint4*)(b);
  f.q[1] = *(const uint4*)(b + 16);
  return f.v;
}

// B-fragment (32x16, col per lane): element e holds K = ((e&8)?e+8:e) + 8*half.
// p points at data contiguous in K -> two contiguous 8-elem (16B) runs at +8*half and +8*half+16.
__device__ __forceinline__ v16bf frag_b_load(const bf16* p, int half) {
  FragU f;
  const bf16* b = p + half * 8;
  f.q[0] = *(const uint4*)(b);
  f.q[1] = *(const uint4*)(b + 16);
  return f.v;
}

// ---------------------------------------------------------------------------
// f32 -> bf16 elementwise convert
// ---------------------------------------------------------------------------
__global__ void cvt_f32_bf16(const float* __restrict__ in, bf16* __restrict__ out, int n) {
  int i = blockIdx.x * 256 + threadIdx.x;
  if (i < n) out[i] = f32_to_bf16(in[i]);
}

// ---------------------------------------------------------------------------
// Tiled bf16 GEMM: C(M x 1024) = A(M x 1024) @ W(1024 x 1024) + bias
// Block tile 128(M) x 64(N), K-step 32. 256 threads = 8 waves in a 4x2 grid,
// each wave computes a 32x32 patch as 2x2 WMMA tiles.
// A tile staged via async-to-LDS; W tile transposed through registers.
// mode 0: out = bf16, per-head layout (B,H,T,64), value scaled by `scale`
// mode 1: out = f32, row-major (M x 1024)
// ---------------------------------------------------------------------------
__global__ void __launch_bounds__(256) gemm_bf16(
    const bf16* __restrict__ A, const bf16* __restrict__ W,
    const float* __restrict__ bias, void* __restrict__ outp,
    int mode, float scale) {
  __shared__ __align__(16) bf16 As[128][40];   // 128x32 tile, padded stride
  __shared__ __align__(16) bf16 Bt[64][40];    // W tile transposed: [n][k]

  const int tid  = threadIdx.x;
  const int wave = tid >> 5;
  const int lane = tid & 31;
  const int half = lane >> 4;
  const int r    = lane & 15;
  const int wm   = wave >> 1;          // 0..3
  const int wn   = wave & 1;           // 0..1
  const int m0   = blockIdx.y * 128;
  const int n0   = blockIdx.x * 64;

  v8f acc[2][2];
  acc[0][0] = {}; acc[0][1] = {}; acc[1][0] = {}; acc[1][1] = {};

  for (int kk = 0; kk < 1024; kk += 32) {
    __syncthreads();
    // A tile: 128 rows x 4 chunks-of-8 = 512 async 16B copies, 2 per thread.
#pragma unroll
    for (int i = 0; i < 2; ++i) {
      int idx = tid + i * 256;
      int row = idx >> 2;
      int c8  = (idx & 3) * 8;
      ASYNC_LDS_B128(A + (size_t)(m0 + row) * 1024 + kk + c8, &As[row][c8]);
    }
    // W tile, transposed into LDS: coalesced global read, scalar LDS scatter.
    {
      int krow = tid >> 3;           // 0..31
      int ng   = (tid & 7) * 8;      // 0..56
      uint4 wv = *(const uint4*)(W + (size_t)(kk + krow) * 1024 + n0 + ng);
      const bf16* wp = (const bf16*)&wv;
#pragma unroll
      for (int e = 0; e < 8; ++e) Bt[ng + e][krow] = wp[e];
      if (kk + 32 < 1024) {
        __builtin_prefetch(A + (size_t)(m0 + (tid >> 1)) * 1024 + kk + 32, 0, 1);
        __builtin_prefetch(W + (size_t)(kk + 32 + krow) * 1024 + n0 + ng, 0, 1);
      }
    }
    __builtin_amdgcn_s_wait_asynccnt(0);
    __syncthreads();

    v16bf a0 = frag_a_load(&As[wm * 32 +  0 + r][0], half);
    v16bf a1 = frag_a_load(&As[wm * 32 + 16 + r][0], half);
    v16bf b0 = frag_b_load(&Bt[wn * 32 +  0 + r][0], half);
    v16bf b1 = frag_b_load(&Bt[wn * 32 + 16 + r][0], half);
    acc[0][0] = WMMA_BF16(a0, b0, acc[0][0]);
    acc[0][1] = WMMA_BF16(a0, b1, acc[0][1]);
    acc[1][0] = WMMA_BF16(a1, b0, acc[1][0]);
    acc[1][1] = WMMA_BF16(a1, b1, acc[1][1]);
  }

#pragma unroll
  for (int ti = 0; ti < 2; ++ti) {
#pragma unroll
    for (int tj = 0; tj < 2; ++tj) {
      int n = n0 + wn * 32 + tj * 16 + r;        // N = lane index
      float bv = bias[n];
#pragma unroll
      for (int i = 0; i < 8; ++i) {
        int m = m0 + wm * 32 + ti * 16 + i + 8 * half;   // M = vgpr + 8*half
        float val = acc[ti][tj][i] + bv;
        if (mode == 0) {
          val *= scale;
          int bidx = m >> 11, t = m & 2047, h = n >> 6, d = n & 63;
          ((bf16*)outp)[(((size_t)(bidx * 16 + h) * 2048 + t) * 64) + d] = f32_to_bf16(val);
        } else {
          ((float*)outp)[(size_t)m * 1024 + n] = val;
        }
      }
    }
  }
}

// ---------------------------------------------------------------------------
// Flash attention (causal, online softmax).
// grid = (T/64 = 32 q-tiles, B*H = 64). 128 threads = 4 waves.
// Each wave owns a 16-row Q strip (frags in registers) and its 64-wide score
// strip; K/V tiles (64 keys) stream through LDS (K via async-to-LDS, V
// transposed through registers). Next tiles prefetched into cache.
// ---------------------------------------------------------------------------
__global__ void __launch_bounds__(128) flash_attn(
    const bf16* __restrict__ Q, const bf16* __restrict__ K,
    const bf16* __restrict__ V, bf16* __restrict__ O) {
  __shared__ __align__(16) bf16 Ks[64][72];        // [key][dim]
  __shared__ __align__(16) bf16 Vt[64][72];        // [dim][key]
  __shared__ __align__(16) bf16 Ps[4][16][72];     // per-wave P strip

  const int bh   = blockIdx.y;
  const int qt   = blockIdx.x;
  const int qb   = qt * 64;
  const int tid  = threadIdx.x;
  const int wave = tid >> 5;
  const int lane = tid & 31;
  const int half = lane >> 4;
  const int r    = lane & 15;

  const bf16* Qp = Q + (size_t)bh * 2048 * 64;
  const bf16* Kp = K + (size_t)bh * 2048 * 64;
  const bf16* Vp = V + (size_t)bh * 2048 * 64;

  // Q strip A-fragments (rows qb + wave*16 + r), resident for whole kernel.
  const int qrow = qb + wave * 16 + r;
  v16bf q0 = frag_a_load(Qp + (size_t)qrow * 64, half);        // d = 0..31
  v16bf q1 = frag_a_load(Qp + (size_t)qrow * 64 + 32, half);   // d = 32..63

  float m_i[8], l_i[8];
  v8f o_acc[4];
#pragma unroll
  for (int i = 0; i < 8; ++i) { m_i[i] = -1e30f; l_i[i] = 0.0f; }
#pragma unroll
  for (int j = 0; j < 4; ++j) o_acc[j] = {};

  for (int kt = 0; kt <= qt; ++kt) {
    const int kb = kt * 64;
    __syncthreads();   // previous iteration done reading Ks/Vt/Ps
    // K tile: async copy row-major. V tile: register transpose into Vt.
#pragma unroll
    for (int i = 0; i < 4; ++i) {
      int idx = tid + i * 128;
      int row = idx >> 3;
      int c8  = (idx & 7) * 8;
      ASYNC_LDS_B128(Kp + (size_t)(kb + row) * 64 + c8, &Ks[row][c8]);
      uint4 vv = *(const uint4*)(Vp + (size_t)(kb + row) * 64 + c8);
      const bf16* vp = (const bf16*)&vv;
#pragma unroll
      for (int e = 0; e < 8; ++e) Vt[c8 + e][row] = vp[e];
      if (kt < qt) {   // prefetch next tile while this one is consumed
        __builtin_prefetch(Kp + (size_t)(kb + 64 + row) * 64 + c8, 0, 1);
        __builtin_prefetch(Vp + (size_t)(kb + 64 + row) * 64 + c8, 0, 1);
      }
    }
    __builtin_amdgcn_s_wait_asynccnt(0);
    __syncthreads();

    // S = Qs @ K^T : B[k=d][n=key] = Ktile[key][d] -> row-major read of Ks.
    v8f s[4];
#pragma unroll
    for (int j = 0; j < 4; ++j) {
      v8f a = {};
      v16bf kb0 = frag_b_load(&Ks[j * 16 + r][0],  half);
      v16bf kb1 = frag_b_load(&Ks[j * 16 + r][32], half);
      a = WMMA_BF16(q0, kb0, a);
      a = WMMA_BF16(q1, kb1, a);
      s[j] = a;
    }

    // Causal mask (only needed on the diagonal tile).
    if (kt == qt) {
#pragma unroll
      for (int j = 0; j < 4; ++j)
#pragma unroll
        for (int i = 0; i < 8; ++i) {
          int rowl = wave * 16 + i + 8 * half;
          int coll = j * 16 + r;
          if (coll > rowl) s[j][i] = -1e30f;
        }
    }

    // Row max: combine 4 tiles in regs, then reduce over 16 lanes of the half.
    float mt[8];
#pragma unroll
    for (int i = 0; i < 8; ++i) {
      float v = fmaxf(fmaxf(s[0][i], s[1][i]), fmaxf(s[2][i], s[3][i]));
#pragma unroll
      for (int off = 1; off <= 8; off <<= 1)
        v = fmaxf(v, __shfl_xor(v, off, 32));
      mt[i] = v;
    }

    float alpha[8];
#pragma unroll
    for (int i = 0; i < 8; ++i) {
      float mn = fmaxf(m_i[i], mt[i]);
      alpha[i] = __expf(m_i[i] - mn);
      m_i[i] = mn;
    }

    // P = exp(S - m), row sums, online rescale of l and O.
    float rs[8];
#pragma unroll
    for (int i = 0; i < 8; ++i) rs[i] = 0.0f;
#pragma unroll
    for (int j = 0; j < 4; ++j)
#pragma unroll
      for (int i = 0; i < 8; ++i) {
        float p = __expf(s[j][i] - m_i[i]);
        s[j][i] = p;
        rs[i] += p;
      }
#pragma unroll
    for (int i = 0; i < 8; ++i) {
      float v = rs[i];
#pragma unroll
      for (int off = 1; off <= 8; off <<= 1)
        v += __shfl_xor(v, off, 32);
      l_i[i] = l_i[i] * alpha[i] + v;
    }
#pragma unroll
    for (int j = 0; j < 4; ++j)
#pragma unroll
      for (int i = 0; i < 8; ++i) o_acc[j][i] *= alpha[i];

    // C-layout -> A-layout conversion for P via LDS.
#pragma unroll
    for (int j = 0; j < 4; ++j)
#pragma unroll
      for (int i = 0; i < 8; ++i)
        Ps[wave][i + 8 * half][j * 16 + r] = f32_to_bf16(s[j][i]);
    __syncthreads();

    // O += P @ V : A = P (K=key), B[k=key][n=d] = Vt[d][key] row read.
    v16bf pa0 = frag_a_load(&Ps[wave][r][0],  half);
    v16bf pa1 = frag_a_load(&Ps[wave][r][32], half);
#pragma unroll
    for (int j = 0; j < 4; ++j) {
      v16bf vb0 = frag_b_load(&Vt[j * 16 + r][0],  half);
      v16bf vb1 = frag_b_load(&Vt[j * 16 + r][32], half);
      o_acc[j] = WMMA_BF16(pa0, vb0, o_acc[j]);
      o_acc[j] = WMMA_BF16(pa1, vb1, o_acc[j]);
    }
  }

  // Epilogue: normalize and store to (B, T, H*Dh) bf16 for the projection GEMM.
  const int b = bh >> 4, h = bh & 15;
#pragma unroll
  for (int i = 0; i < 8; ++i) {
    float inv = 1.0f / l_i[i];
    int tq = qb + wave * 16 + i + 8 * half;
    size_t base = ((size_t)(b * 2048 + tq)) * 1024 + (size_t)h * 64;
#pragma unroll
    for (int j = 0; j < 4; ++j)
      O[base + j * 16 + r] = f32_to_bf16(o_acc[j][i] * inv);
  }
}

// ---------------------------------------------------------------------------
// Launch: convert -> QKV GEMMs -> flash attention -> output projection.
// Workspace usage: 16 + 4*2 + 3*16 + 16 = 88 MiB.
// ---------------------------------------------------------------------------
extern "C" void kernel_launch(void* const* d_in, const int* in_sizes, int n_in,
                              void* d_out, int out_size, void* d_ws, size_t ws_size,
                              hipStream_t stream) {
  (void)in_sizes; (void)n_in; (void)out_size; (void)ws_size;
  const float* x  = (const float*)d_in[0];
  const float* Wk = (const float*)d_in[1];
  const float* bk = (const float*)d_in[2];
  const float* Wq = (const float*)d_in[3];
  const float* bq = (const float*)d_in[4];
  const float* Wv = (const float*)d_in[5];
  const float* bv = (const float*)d_in[6];
  const float* Wp = (const float*)d_in[7];
  const float* bp = (const float*)d_in[8];
  float* out = (float*)d_out;

  const size_t MT = 8192;   // B*T
  const size_t DM = 1024;

  char* ws = (char*)d_ws;
  bf16* xb  = (bf16*)ws; ws += MT * DM * sizeof(bf16);
  bf16* wqb = (bf16*)ws; ws += DM * DM * sizeof(bf16);
  bf16* wkb = (bf16*)ws; ws += DM * DM * sizeof(bf16);
  bf16* wvb = (bf16*)ws; ws += DM * DM * sizeof(bf16);
  bf16* wpb = (bf16*)ws; ws += DM * DM * sizeof(bf16);
  bf16* Qb  = (bf16*)ws; ws += MT * DM * sizeof(bf16);
  bf16* Kb  = (bf16*)ws; ws += MT * DM * sizeof(bf16);
  bf16* Vb  = (bf16*)ws; ws += MT * DM * sizeof(bf16);
  bf16* Ab  = (bf16*)ws; ws += MT * DM * sizeof(bf16);

  cvt_f32_bf16<<<(int)((MT * DM) / 256), 256, 0, stream>>>(x,  xb,  (int)(MT * DM));
  cvt_f32_bf16<<<(int)((DM * DM) / 256), 256, 0, stream>>>(Wq, wqb, (int)(DM * DM));
  cvt_f32_bf16<<<(int)((DM * DM) / 256), 256, 0, stream>>>(Wk, wkb, (int)(DM * DM));
  cvt_f32_bf16<<<(int)((DM * DM) / 256), 256, 0, stream>>>(Wv, wvb, (int)(DM * DM));
  cvt_f32_bf16<<<(int)((DM * DM) / 256), 256, 0, stream>>>(Wp, wpb, (int)(DM * DM));

  dim3 ggrid(16, 64);  // N/64, M/128
  // Q is pre-scaled by 1/sqrt(Dh) = 0.125
  gemm_bf16<<<ggrid, 256, 0, stream>>>(xb, wqb, bq, (void*)Qb, 0, 0.125f);
  gemm_bf16<<<ggrid, 256, 0, stream>>>(xb, wkb, bk, (void*)Kb, 0, 1.0f);
  gemm_bf16<<<ggrid, 256, 0, stream>>>(xb, wvb, bv, (void*)Vb, 0, 1.0f);

  flash_attn<<<dim3(32, 64), 128, 0, stream>>>(Qb, Kb, Vb, Ab);

  gemm_bf16<<<ggrid, 256, 0, stream>>>(Ab, wpb, bp, (void*)out, 1, 1.0f);
}
